// AGREE_3367254360325
// MI455X (gfx1250) — matrix operand order repaired and checked by reading, not compiled, not run on hardware
//
#include <hip/hip_runtime.h>
#include <math.h>

typedef __attribute__((ext_vector_type(16))) _Float16 v16h;
typedef __attribute__((ext_vector_type(8)))  float    v8f;

#define BATCH 16384
#define MEMB  50
#define DIM   64

// One workgroup (128 threads = 4 wave32) per batch row. Fully fused:
//   gather -> WMMA attention scores -> masked softmax -> weighted sum -> predict MLP.
// A and B f16 operands are stored in LDS pre-swizzled into the exact WMMA lane
// layout so each lane's operand is a contiguous 32B LDS load (2x ds_load_b128).
__global__ __launch_bounds__(128) void agree_fused_kernel(
    const int*   __restrict__ group_inputs,   // [B]
    const int*   __restrict__ item_inputs,    // [B]
    const int*   __restrict__ member_ids,     // [B,M]
    const int*   __restrict__ member_lengths, // [B]
    const float* __restrict__ user_table,     // [NU,D]
    const float* __restrict__ item_table,     // [NI,D]
    const float* __restrict__ group_table,    // [NG,D]
    const float* __restrict__ att_w1,         // [2D,16]
    const float* __restrict__ att_b1,         // [16]
    const float* __restrict__ att_w2,         // [16,1]
    const float* __restrict__ att_b2,         // [1]
    const float* __restrict__ pred_w1,        // [3D,8]
    const float* __restrict__ pred_b1,        // [8]
    const float* __restrict__ pred_w2,        // [8,1]
    const float* __restrict__ pred_b2,        // [1]
    float*       __restrict__ out)            // [B,1]
{
    __shared__ float    s_mem[MEMB][DIM];   // f32 member embeddings (exact path)
    // A operand, pre-swizzled: [kb(2)][row(64)][khalf(2)] -> 16 contiguous halves
    __shared__ v16h     s_Av[2 * 64 * 2];
    // B operand, pre-swizzled: [kb(2)][lane(32)] -> 16 contiguous halves
    __shared__ v16h     s_Bv[2 * 32];
    __shared__ float    s_item[DIM];
    __shared__ float    s_group[DIM];
    __shared__ float    s_iconst[16];       // b1 + item @ W1[64:] (folded into C)
    __shared__ float4   s_hp4[64 * 4];      // relu(H)*w2 partials, [row][16 n]
    __shared__ float    s_scores[64];
    __shared__ float    s_weights[64];
    __shared__ int      s_ids[MEMB];
    __shared__ float4   s_red4[8 * 16];     // predict-MLP partials [j][64 d]
    __shared__ float    s_ph[8];

    _Float16* s_A     = (_Float16*)s_Av;
    _Float16* s_B     = (_Float16*)s_Bv;
    float*    s_hp    = (float*)s_hp4;
    float*    s_red   = (float*)s_red4;

    const int b = blockIdx.x;
    const int t = threadIdx.x;

    // ---- stage indices, item/group embeddings, swizzled f16 W1, zero-pad ----
    if (t < MEMB) s_ids[t] = member_ids[b * MEMB + t];
    if (t < DIM)  s_item[t]       = item_table [(long)item_inputs [b] * DIM + t];
    else          s_group[t - 64] = group_table[(long)group_inputs[b] * DIM + (t - 64)];
    // B swizzle: K = kb*32 + (l>>4)*8 + (h<8 ? h : h+8) ; N = l&15
    for (int i = t; i < 2 * 32 * 16; i += 128) {
        int kb = i >> 9, l = (i >> 4) & 31, h = i & 15;
        int k = kb * 32 + ((l >> 4) << 3) + (h < 8 ? h : h + 8);
        s_B[i] = (_Float16)att_w1[k * 16 + (l & 15)];
    }
    // zero A rows 50..63 (two contiguous half-ranges, one per kb)
    for (int i = t; i < 2 * (64 - MEMB) * 32; i += 128) {
        int idx = (i < 448) ? (MEMB * 32 + i) : ((64 + MEMB) * 32 + (i - 448));
        s_A[idx] = (_Float16)0.f;
    }
    __syncthreads();

    // ---- gather member embeddings (float4, L2-resident tables) ----
    // A swizzle: k = kb*32 + khalf*8 + (h<8 ? h : h+8)  =>
    //   rem=k&31: khalf=(rem>>3)&1, h=(rem&7)+((rem>>4)<<3); 4-chunks never cross a run.
    const float4* ut4 = (const float4*)user_table;
    for (int i = t; i < MEMB * (DIM / 4); i += 128) {
        int m = i >> 4, c = i & 15;
        float4 v = ut4[(long)s_ids[m] * (DIM / 4) + c];
        int d = c * 4;
        s_mem[m][d + 0] = v.x; s_mem[m][d + 1] = v.y;
        s_mem[m][d + 2] = v.z; s_mem[m][d + 3] = v.w;
        int kb = d >> 5, rem = d & 31;
        int khalf = (rem >> 3) & 1;
        int h0 = (rem & 7) + ((rem >> 4) << 3);
        _Float16* dst = s_A + (((kb * 64 + m) * 2 + khalf) * 16 + h0);
        dst[0] = (_Float16)v.x; dst[1] = (_Float16)v.y;
        dst[2] = (_Float16)v.z; dst[3] = (_Float16)v.w;
    }
    // item-constant part of attention hidden layer: iconst[j] = b1[j] + item @ W1[64:,j]
    if (t < 16) {
        float acc = att_b1[t];
        for (int k = 0; k < DIM; ++k) acc += s_item[k] * att_w1[(DIM + k) * 16 + t];
        s_iconst[t] = acc;
    }
    __syncthreads();

    // ---- WMMA attention: H = relu(mem @ W1[:64] + iconst), scores = H @ w2 + b2 ----
    const int wave  = t >> 5, lane = t & 31;
    const int mrow  = wave * 16 + (lane & 15);
    const int khalf = lane >> 4;
    v16h a0 = s_Av[(0 * 64 + mrow) * 2 + khalf];
    v16h a1 = s_Av[(1 * 64 + mrow) * 2 + khalf];
    v16h b0 = s_Bv[lane];
    v16h b1 = s_Bv[32 + lane];
    float ic = s_iconst[lane & 15];
    v8f c = {ic, ic, ic, ic, ic, ic, ic, ic};
    c = __builtin_amdgcn_wmma_f32_16x16x32_f16(false, a0, false, b0, (short)0, c, false, false);
    c = __builtin_amdgcn_wmma_f32_16x16x32_f16(false, a1, false, b1, (short)0, c, false, false);

    // relu + scale by w2[n], scatter partials to [row][n]; reduce per row after barrier
    float w2n = att_w2[lane & 15];
    #pragma unroll
    for (int r = 0; r < 8; ++r) {
        int rowd = wave * 16 + khalf * 8 + r;   // C/D layout: VGPR r -> M = r + 8*(lane/16)
        s_hp[rowd * 16 + (lane & 15)] = fmaxf(c[r], 0.f) * w2n;
    }
    __syncthreads();
    if (t < 64) {
        float4 p0 = s_hp4[t * 4 + 0], p1 = s_hp4[t * 4 + 1];
        float4 p2 = s_hp4[t * 4 + 2], p3 = s_hp4[t * 4 + 3];
        float s = ((p0.x + p0.y) + (p0.z + p0.w)) + ((p1.x + p1.y) + (p1.z + p1.w))
                + ((p2.x + p2.y) + (p2.z + p2.w)) + ((p3.x + p3.y) + (p3.z + p3.w));
        s_scores[t] = s + att_b2[0];
    }
    __syncthreads();

    // ---- masked softmax over 50 members (single wave) ----
    if (t < 32) {
        int  len = member_lengths[b];
        int  m1  = t + 32;
        bool v0  = (t  < MEMB) && (t  <= len);
        bool v1  = (m1 < MEMB) && (m1 <= len);
        float x0 = v0 ? s_scores[t]  : -INFINITY;
        float x1 = v1 ? s_scores[m1] : -INFINITY;
        float mx = fmaxf(x0, x1);
        for (int o = 16; o > 0; o >>= 1) mx = fmaxf(mx, __shfl_xor(mx, o, 32));
        float e0 = v0 ? __expf(x0 - mx) : 0.f;
        float e1 = v1 ? __expf(x1 - mx) : 0.f;
        float sm = e0 + e1;
        for (int o = 16; o > 0; o >>= 1) sm += __shfl_xor(sm, o, 32);
        float inv = 1.f / sm;   // at least one valid member (m=0 always <= len)
        s_weights[t]  = e0 * inv;
        s_weights[m1] = e1 * inv;
    }
    __syncthreads();

    // ---- weighted member sum + fused predict-MLP partials (f32 path) ----
    if (t < DIM) {
        float acc = 0.f;
        for (int m = 0; m < MEMB; ++m) acc += s_weights[m] * s_mem[m][t];
        float gd = acc + s_group[t];
        float it = s_item[t];
        float el = gd * it;
        #pragma unroll
        for (int j = 0; j < 8; ++j)
            s_red[j * DIM + t] = el * pred_w1[t * 8 + j]
                               + gd * pred_w1[(64 + t) * 8 + j]
                               + it * pred_w1[(128 + t) * 8 + j];
    }
    __syncthreads();
    if (t < 8) {
        float acc = pred_b1[t];
        #pragma unroll
        for (int q = 0; q < 16; ++q) {
            float4 v = s_red4[t * 16 + q];
            acc += (v.x + v.y) + (v.z + v.w);
        }
        s_ph[t] = fmaxf(acc, 0.f);
    }
    __syncthreads();
    if (t == 0) {
        float z = pred_b2[0];
        for (int j = 0; j < 8; ++j) z += s_ph[j] * pred_w2[j];
        out[b] = 1.f / (1.f + __expf(-z));
    }
}

extern "C" void kernel_launch(void* const* d_in, const int* in_sizes, int n_in,
                              void* d_out, int out_size, void* d_ws, size_t ws_size,
                              hipStream_t stream) {
    (void)in_sizes; (void)n_in; (void)out_size; (void)d_ws; (void)ws_size;
    agree_fused_kernel<<<BATCH, 128, 0, stream>>>(
        (const int*)  d_in[0],  // group_inputs
        (const int*)  d_in[1],  // item_inputs
        (const int*)  d_in[2],  // member_ids
        (const int*)  d_in[3],  // member_lengths
        (const float*)d_in[4],  // user_table
        (const float*)d_in[5],  // item_table
        (const float*)d_in[6],  // group_table
        (const float*)d_in[7],  // att_w1
        (const float*)d_in[8],  // att_b1
        (const float*)d_in[9],  // att_w2
        (const float*)d_in[10], // att_b2
        (const float*)d_in[11], // pred_w1
        (const float*)d_in[12], // pred_b1
        (const float*)d_in[13], // pred_w2
        (const float*)d_in[14], // pred_b2
        (float*)d_out);
}
